// BatchedGATWrapper_52080773431339
// MI455X (gfx1250) — matrix operand order repaired and checked by reading, not compiled, not run on hardware
//
#include <hip/hip_runtime.h>
#include <hip/hip_bf16.h>

// ---------------------------------------------------------------------------
// Problem constants (match reference)
// ---------------------------------------------------------------------------
#define Bn 8
#define Nn 1024
#define Dn 512
#define Hn 8
#define Cn 64
#define HC 512   // H*C
#define LEAKY 0.2f

typedef __attribute__((ext_vector_type(2)))  float  v2f;
typedef __attribute__((ext_vector_type(8)))  float  v8f;
typedef __attribute__((ext_vector_type(16))) __bf16 v16bf;

__device__ __forceinline__ unsigned short f32_to_bf16_bits(float f) {
    union { float f; unsigned u; } x; x.f = f;
    unsigned r = x.u + 0x7FFFu + ((x.u >> 16) & 1u);   // round-to-nearest-even
    return (unsigned short)(r >> 16);
}

// ---------------------------------------------------------------------------
// Kernel 1: projection  x = features @ W   (f32 WMMA 16x16x4)
//   Each wave computes a 16-row x 64-col strip (one head): the A operand is
//   loaded once per K-step and reused by 4 WMMAs (4x less A traffic), and the
//   4 B loads per K-row form one contiguous 256B stream.
//   Also emits bf16 transposed copy xt[b][h][c][n] for the aggregation GEMM.
// ---------------------------------------------------------------------------
__global__ __launch_bounds__(128) void gat_proj_kernel(
    const float* __restrict__ feat,   // [B*N, D]
    const float* __restrict__ W,      // [D, HC]
    float*       __restrict__ x,      // [B*N, HC]  (ws)
    unsigned short* __restrict__ xt)  // [B, H, C, N] bf16 (ws)
{
    const int lane = threadIdx.x & 31;
    const int wave = threadIdx.x >> 5;
    const int r    = lane & 15;       // row within tile (A), col within strip (B/C)
    const int half = lane >> 4;       // 0: K lo pair, 1: K hi pair

    const long row0    = (long)blockIdx.x * 16;          // flattened B*N row tile
    const int  colBase = (blockIdx.y * 4 + wave) * 64;   // 64-col strip == one head

    const float* Arow = feat + (row0 + r) * Dn;

    v8f acc[4] = {{}, {}, {}, {}};
    // A 16x4 f32 layout: lanes 0-15 M=0-15 {V0=K0,V1=K1}; lanes 16-31 {V0=K2,V1=K3}
    // B 4x16 f32 layout (mirror): V0 = K0(lo lanes)/K2(hi lanes), V1 = K1/K3
    for (int k = 0; k < Dn; k += 4) {
        v2f a = *reinterpret_cast<const v2f*>(Arow + k + half * 2);
        const float* w0 = W + (k + half * 2 + 0) * HC + colBase + r;
        const float* w1 = W + (k + half * 2 + 1) * HC + colBase + r;
        #pragma unroll
        for (int t = 0; t < 4; ++t) {
            v2f bop;
            bop.x = w0[t * 16];
            bop.y = w1[t * 16];
            acc[t] = __builtin_amdgcn_wmma_f32_16x16x4_f32(
                         false, a, false, bop, (short)0, acc[t], false, false);
        }
    }

    // C/D layout: VGPR g holds M = g + (lane<16 ? 0 : 8), N = lane%16
    const int  h    = colBase >> 6;              // whole strip is one head
    const long bidx = row0 / Nn;                 // tiles never straddle batches
    const long n0   = (row0 % Nn) + (half ? 8 : 0);

    #pragma unroll
    for (int t = 0; t < 4; ++t) {
        const int col = colBase + t * 16 + r;
        const int cc  = t * 16 + r;              // c within head
        unsigned short pk[8];
        #pragma unroll
        for (int g = 0; g < 8; ++g) {
            long gr = row0 + g + (half ? 8 : 0);
            float v = acc[t][g];
            x[gr * HC + col] = v;                // f32 copy (for logits kernel)
            pk[g] = f32_to_bf16_bits(v);
        }
        uint4 packed;
        packed.x = (unsigned)pk[0] | ((unsigned)pk[1] << 16);
        packed.y = (unsigned)pk[2] | ((unsigned)pk[3] << 16);
        packed.z = (unsigned)pk[4] | ((unsigned)pk[5] << 16);
        packed.w = (unsigned)pk[6] | ((unsigned)pk[7] << 16);
        long xtIdx = ((((long)bidx * Hn + h) * Cn + cc) * Nn) + n0;
        *reinterpret_cast<uint4*>(xt + xtIdx) = packed;   // 16B aligned
    }
}

// ---------------------------------------------------------------------------
// Kernel 2: per-node attention logits  a_src/a_dst[b,h,n] (f32, from f32 x)
// ---------------------------------------------------------------------------
__global__ __launch_bounds__(256) void gat_logits_kernel(
    const float* __restrict__ x,        // [B*N, HC]
    const float* __restrict__ att_src,  // [H, C]
    const float* __restrict__ att_dst,  // [H, C]
    float* __restrict__ asrcT,          // [B, H, N] (ws)
    float* __restrict__ adstT)          // [B, H, N] (ws)
{
    int t  = blockIdx.x * 256 + threadIdx.x;    // [0, B*H*N)
    int n  = t & (Nn - 1);
    int bh = t >> 10;
    int h  = bh & (Hn - 1);
    int b  = bh >> 3;

    const float* xr = x + ((long)(b * Nn + n)) * HC + h * Cn;
    const float* as = att_src + h * Cn;
    const float* ad = att_dst + h * Cn;
    float s = 0.f, d = 0.f;
    #pragma unroll
    for (int c = 0; c < Cn; c += 4) {
        float4 xv = *reinterpret_cast<const float4*>(xr + c);
        float4 sv = *reinterpret_cast<const float4*>(as + c);
        float4 dv = *reinterpret_cast<const float4*>(ad + c);
        s += xv.x * sv.x + xv.y * sv.y + xv.z * sv.z + xv.w * sv.w;
        d += xv.x * dv.x + xv.y * dv.y + xv.z * dv.z + xv.w * dv.w;
    }
    asrcT[t] = s;
    adstT[t] = d;
}

// ---------------------------------------------------------------------------
// Kernel 3: fused masked softmax + aggregation (flash-attention style)
//   Block = 256 threads (8 waves) per (b, 16-row i-tile); wave w = head w.
//   a_src staged into LDS with GLOBAL_LOAD_ASYNC_TO_LDS_B128 (ASYNCcnt),
//   overlapped with the adjacency-bitmask build, fenced by s_wait_asynccnt.
// ---------------------------------------------------------------------------
__global__ __launch_bounds__(256) void gat_attn_kernel(
    const int*   __restrict__ adj,     // [B, N, N]  (edge j->i is adj[b,j,i])
    const float* __restrict__ asrcT,   // [B, H, N]
    const float* __restrict__ adstT,   // [B, H, N]
    const unsigned short* __restrict__ xt,  // [B, H, C, N] bf16
    const float* __restrict__ bias,    // [HC]
    float*       __restrict__ out)     // [B, N, HC]
{
    __shared__ __align__(16) unsigned smask[Nn];      // bit i of smask[j] = edge j->(i0+i)
    __shared__ __align__(16) float    sasrc[Hn * Nn]; // a_src all heads (32KB)

    const int b  = blockIdx.x >> 6;       // N/16 = 64 i-tiles
    const int i0 = (blockIdx.x & 63) * 16;

    // ---- phase 0a: kick off async global->LDS copy of a_src (8 x b128/thread)
    {
        const float* gsrc = asrcT + (long)b * Hn * Nn;
        #pragma unroll
        for (int it = 0; it < (Hn * Nn) / (4 * 256); ++it) {   // 8 iters
            int idx = (it * 256 + threadIdx.x) * 4;            // float4 index
            unsigned lds_off = (unsigned)(uintptr_t)(&sasrc[idx]);  // addr[31:0] == LDS addr
            unsigned long long gaddr = (unsigned long long)(uintptr_t)(gsrc + idx);
            asm volatile("global_load_async_to_lds_b128 %0, %1, off"
                         :: "v"(lds_off), "v"(gaddr) : "memory");
        }
    }

    // ---- phase 0b: build mask bits (overlaps with async copy) ----
    for (int j = threadIdx.x; j < Nn; j += 256) {
        const int* arow = adj + ((long)(b * Nn + j)) * Nn + i0;  // 16 contiguous ints
        unsigned msk = 0;
        #pragma unroll
        for (int q = 0; q < 16; q += 4) {
            int4 av = *reinterpret_cast<const int4*>(arow + q);
            if (av.x) msk |= 1u << (q + 0);
            if (av.y) msk |= 1u << (q + 1);
            if (av.z) msk |= 1u << (q + 2);
            if (av.w) msk |= 1u << (q + 3);
        }
        if (j >= i0 && j < i0 + 16) msk |= 1u << (j - i0);       // self-loop
        smask[j] = msk;
    }

    asm volatile("s_wait_asynccnt 0x0" ::: "memory");   // async LDS data landed
    __syncthreads();

    const int h    = threadIdx.x >> 5;    // wave id == head
    const int lane = threadIdx.x & 31;
    const int r    = lane & 15;
    const int half = lane >> 4;
    const float* as = &sasrc[h * Nn];

    const float a_d = adstT[((long)(b * Hn + h)) * Nn + i0 + r];

    // ---- phase 1: online max/sum over sources j (2 lanes per row) ----
    float m = -3.0e38f, s = 0.f;
    for (int j = half; j < Nn; j += 2) {
        if ((smask[j] >> r) & 1u) {
            float e = a_d + as[j];
            e = fmaxf(e, LEAKY * e);                 // leaky relu (slope<1)
            float mn = fmaxf(m, e);
            s = s * __expf(m - mn) + __expf(e - mn);
            m = mn;
        }
    }
    float m2 = __shfl_xor(m, 16, 32);
    float s2 = __shfl_xor(s, 16, 32);
    float mm = fmaxf(m, m2);
    s = s * __expf(m - mm) + s2 * __expf(m2 - mm);
    m = mm;
    const float inv_s = 1.0f / s;                    // s>0 (self-loop guarantees)

    // ---- phase 2: P(bf16) @ V(bf16) with WMMA f32 16x16x32 ----
    const unsigned short* xth = xt + ((long)(b * Hn + h)) * Cn * Nn;
    v8f acc[4] = {{}, {}, {}, {}};

    for (int jb = 0; jb < Nn; jb += 32) {
        // A tile 16x32 bf16 in-register (contiguous k order, same as B):
        // lane<16 holds row r, K = 0..15; lane>=16: K = 16..31
        v16bf amat;
        #pragma unroll
        for (int e = 0; e < 16; ++e) {
            int k  = e + half * 16;
            int j  = jb + k;
            float al = 0.f;
            if ((smask[j] >> r) & 1u) {
                float ev = a_d + as[j];
                ev = fmaxf(ev, LEAKY * ev);
                al = __expf(ev - m) * inv_s;
            }
            unsigned short bits = f32_to_bf16_bits(al);
            amat[e] = *reinterpret_cast<__bf16*>(&bits);
        }
        // B tiles: 32x16 bf16; lane<16 -> K=0..15 (32B contiguous), lane>=16 -> K=16..31
        #pragma unroll
        for (int t = 0; t < 4; ++t) {
            const v16bf bmat = *reinterpret_cast<const v16bf*>(
                xth + (long)(t * 16 + r) * Nn + jb + half * 16);
            acc[t] = __builtin_amdgcn_wmma_f32_16x16x32_bf16(
                         false, amat, false, bmat, (short)0, acc[t], false, false);
        }
    }

    // ---- epilogue: bias + ELU, store [B,N,HC] ----
    #pragma unroll
    for (int t = 0; t < 4; ++t) {
        int   col = h * Cn + t * 16 + r;
        float bi  = bias[col];
        #pragma unroll
        for (int g = 0; g < 8; ++g) {
            int   row = i0 + g + (half ? 8 : 0);
            float v   = acc[t][g] + bi;
            v = (v > 0.f) ? v : (__expf(v) - 1.f);   // ELU (alpha=1)
            out[((long)(b * Nn + row)) * HC + col] = v;
        }
    }
}

// ---------------------------------------------------------------------------
// Launch
// ---------------------------------------------------------------------------
extern "C" void kernel_launch(void* const* d_in, const int* in_sizes, int n_in,
                              void* d_out, int out_size, void* d_ws, size_t ws_size,
                              hipStream_t stream) {
    const float* feat    = (const float*)d_in[0];  // [B,N,D]
    const int*   adj     = (const int*)  d_in[1];  // [B,N,N]
    const float* W       = (const float*)d_in[2];  // [D,HC]
    const float* att_src = (const float*)d_in[3];  // [H,C]
    const float* att_dst = (const float*)d_in[4];  // [H,C]
    const float* bias    = (const float*)d_in[5];  // [HC]
    float*       out     = (float*)d_out;          // [B,N,HC]

    char* ws = (char*)d_ws;
    float*          x_f32 = (float*)(ws);                              // 16 MB
    unsigned short* xt    = (unsigned short*)(ws + 16777216);          //  8 MB
    float*          asrcT = (float*)(ws + 16777216 + 8388608);         // 256 KB
    float*          adstT = (float*)(ws + 16777216 + 8388608 + 262144);

    dim3 g1((Bn * Nn) / 16, HC / 256);  // (512, 2), 4 waves x 64-col strips
    gat_proj_kernel<<<g1, 128, 0, stream>>>(feat, W, x_f32, xt);

    gat_logits_kernel<<<(Bn * Hn * Nn) / 256, 256, 0, stream>>>(
        x_f32, att_src, att_dst, asrcT, adstT);

    gat_attn_kernel<<<Bn * (Nn / 16), 256, 0, stream>>>(
        adj, asrcT, adstT, xt, bias, out);
}
